// ScaledDotProductAttention_14912126452530
// MI455X (gfx1250) — matrix operand order repaired and checked by reading, not compiled, not run on hardware
//
#include <hip/hip_runtime.h>

// Problem constants (from reference)
#define BATCH 16
#define NQ_   512
#define NK_   1024
#define HEADS 8
#define LD    1024   // row stride (elements) of all projected tensors

typedef __attribute__((ext_vector_type(16))) __bf16 v16bf;
typedef __attribute__((ext_vector_type(8)))  float  v8f;
typedef __attribute__((ext_vector_type(4)))  unsigned int u32x4;
typedef __attribute__((ext_vector_type(8)))  int i32x8;
typedef __attribute__((ext_vector_type(4)))  int i32x4;

#if __has_builtin(__builtin_amdgcn_tensor_load_to_lds)
#define HAVE_TDM 1
#else
#define HAVE_TDM 0
#endif

union V16U { v16bf v; unsigned short s[16]; uint4 q[2]; };

__device__ __forceinline__ unsigned short f2bf(float f) {
  unsigned u = __builtin_bit_cast(unsigned, f);
  u += 0x7FFFu + ((u >> 16) & 1u);          // round-to-nearest-even
  return (unsigned short)(u >> 16);
}

// pack two floats -> two bf16 in one dword (hardware cvt if available)
__device__ __forceinline__ unsigned pk2bf(float lo, float hi) {
#if __has_builtin(__builtin_amdgcn_cvt_pk_bf16_f32)
  auto r = __builtin_amdgcn_cvt_pk_bf16_f32(lo, hi);
  return __builtin_bit_cast(unsigned, r);
#else
  return ((unsigned)f2bf(hi) << 16) | (unsigned)f2bf(lo);
#endif
}

// A-fragment (16x32 bf16): two contiguous 16B chunks per lane
__device__ __forceinline__ v16bf ld_a_bf16(const unsigned short* p0,
                                           const unsigned short* p1) {
  V16U r;
  r.q[0] = *(const uint4*)p0;
  r.q[1] = *(const uint4*)p1;
  return r.v;
}

// B-fragment (32x16 bf16): one contiguous 32B chunk per lane
__device__ __forceinline__ v16bf ld_b_bf16(const unsigned short* p) {
  V16U r;
  r.q[0] = *(const uint4*)p;
  r.q[1] = *(const uint4*)(p + 8);
  return r.v;
}

#define WMMA_BF16(a, b, c) \
  __builtin_amdgcn_wmma_f32_16x16x32_bf16(false, (a), false, (b), (short)0, (c), false, false)

// ---------------------------------------------------------------------------
// Elementwise f32 -> bf16 (one-time precast of activations & weights).
// ---------------------------------------------------------------------------
__global__ __launch_bounds__(256)
void cvt_bf16_kernel(const float* __restrict__ in,
                     unsigned short* __restrict__ out, int n) {
  const int i = (blockIdx.x * 256 + threadIdx.x) * 8;
  if (i >= n) return;
  float4 a = *(const float4*)(in + i);
  float4 b = *(const float4*)(in + i + 4);
  uint4 r;
  r.x = pk2bf(a.x, a.y);
  r.y = pk2bf(a.z, a.w);
  r.z = pk2bf(b.x, b.y);
  r.w = pk2bf(b.z, b.w);
  *(uint4*)(out + i) = r;
}

// ---------------------------------------------------------------------------
// Projection: Y[M,1024](bf16) = X[M,128](bf16) @ W[1024,128]^T + bias(f32)
// WG = 128 thr = 4 waves; wave tile = 16 rows x 64 cols; 16 WMMA / wave.
// ---------------------------------------------------------------------------
__global__ __launch_bounds__(128)
void proj_kernel(const unsigned short* __restrict__ X,
                 const unsigned short* __restrict__ W,
                 const float* __restrict__ bias, unsigned short* __restrict__ Y,
                 int M) {
  const int tid  = threadIdx.x;
  const int wave = tid >> 5, lane = tid & 31;
  const int n = lane & 15, hf = lane >> 4;
  const int rblocks = M >> 6;
  const int rb = blockIdx.x % rblocks;
  const int nb = blockIdx.x / rblocks;
  const int arow = rb * 64 + wave * 16 + n;
  const int col0 = nb * 64;

  const unsigned short* xrow = X + (size_t)arow * 128;
  v8f z = {0.f, 0.f, 0.f, 0.f, 0.f, 0.f, 0.f, 0.f};
  v8f acc[4] = {z, z, z, z};

#pragma unroll
  for (int c = 0; c < 4; ++c) {
    v16bf a = ld_a_bf16(xrow + c * 32 + hf * 16, xrow + c * 32 + hf * 16 + 8);
#pragma unroll
    for (int t = 0; t < 4; ++t) {
      const unsigned short* wrow = W + (size_t)(col0 + t * 16 + n) * 128;
      v16bf bfr = ld_b_bf16(wrow + c * 32 + hf * 16);
      acc[t] = WMMA_BF16(a, bfr, acc[t]);
    }
  }
#pragma unroll
  for (int t = 0; t < 4; ++t) {
    const int col = col0 + t * 16 + n;
    const float bv = bias[col];
#pragma unroll
    for (int r = 0; r < 8; ++r) {
      const int orow = rb * 64 + wave * 16 + r + 8 * hf;
      Y[(size_t)orow * LD + col] = f2bf(acc[t][r] + bv);
    }
  }
}

// ---------------------------------------------------------------------------
// Flash attention (row softmax): O = softmax_row(Q K^T * scale) V
// K tile staged via the Tensor Data Mover (TDM) when available; V staged
// manually (transposed). WG = 4 waves, 64 q-rows / WG. Key blocks of 32.
// ---------------------------------------------------------------------------
__global__ __launch_bounds__(128)
void flash_kernel(const unsigned short* __restrict__ Q,
                  const unsigned short* __restrict__ K,
                  const unsigned short* __restrict__ V,
                  unsigned short* __restrict__ O,
                  int MQ, int MK) {
  __shared__ unsigned short Kt[32 * 128];   // [key][kd]   8KB
  __shared__ unsigned short Vt[128 * 32];   // [vd][key]   8KB (transposed)
  __shared__ unsigned short Ps[4][16 * 32]; // per-wave P scratch, 4KB

  const int tid  = threadIdx.x;
  const int wave = tid >> 5, lane = tid & 31;
  const int n = lane & 15, hf = lane >> 4;
  const int qblocks = MQ >> 6;
  const int qb = blockIdx.x % qblocks;
  const int h  = (blockIdx.x / qblocks) & 7;
  const int b  = blockIdx.x / (qblocks * 8);

  const unsigned short* qrow =
      Q + (size_t)(b * MQ + qb * 64 + wave * 16 + n) * LD + h * 128;

  v16bf qf[4];
#pragma unroll
  for (int c = 0; c < 4; ++c)
    qf[c] = ld_a_bf16(qrow + c * 32 + hf * 16, qrow + c * 32 + hf * 16 + 8);

  v8f z = {0.f, 0.f, 0.f, 0.f, 0.f, 0.f, 0.f, 0.f};
  v8f o[8] = {z, z, z, z, z, z, z, z};
  float m[8], l[8];
#pragma unroll
  for (int r = 0; r < 8; ++r) { m[r] = -1e30f; l[r] = 0.f; }

  const float scale = 0.0883883476483184f; // 1/sqrt(128)
  const int cr = tid >> 2, cc = (tid & 3) * 32; // cooperative-load coords

#if HAVE_TDM
  const unsigned ldsK = (unsigned)(size_t)(void*)&Kt[0];
#endif

  for (int kb = 0; kb < MK; kb += 32) {
    __syncthreads();
    // ---- stage K tile [32 x 128] row-major into LDS ----
#if HAVE_TDM
    if (wave == 0) {
      // 2-D Tensor DMA descriptor (CDNA5 ISA)
      const unsigned long long ga =
          (unsigned long long)(const void*)(K + (size_t)(b * MK + kb) * LD + h * 128);
      u32x4 g0;
      g0[0] = 1u;                                   // count=1, user desc
      g0[1] = ldsK;                                 // lds_addr (bytes)
      g0[2] = (unsigned)ga;                         // global_addr[31:0]
      g0[3] = (unsigned)(ga >> 32) | (2u << 30);    // global_addr[56:32] | type=2
      i32x8 g1;
      g1[0] = 0x00010000;                           // data_size=1 (2B), mask=0
      g1[1] = (int)(1024u << 16);                   // tensor_dim0 = 1024 (lo16)
      g1[2] = (int)(32u << 16);                     // tensor_dim0 hi=0 | tensor_dim1=32
      g1[3] = (int)(128u << 16);                    // tensor_dim1 hi=0 | tile_dim0=128
      g1[4] = 32;                                   // tile_dim1=32, tile_dim2=0
      g1[5] = 1024;                                 // tensor_dim0_stride = 1024
      g1[6] = 0;
      g1[7] = 0;
      i32x4 zg4 = {0, 0, 0, 0};                     // groups 2/3 unused (2-D tile)
      i32x8 zg8 = {0, 0, 0, 0, 0, 0, 0, 0};
      __builtin_amdgcn_tensor_load_to_lds(g0, g1, zg4, zg4, zg8, 0);
    }
#else
    {
      const unsigned short* krow =
          K + (size_t)(b * MK + kb + cr) * LD + h * 128 + cc;
      uint4* kdst = (uint4*)&Kt[cr * 128 + cc];
#pragma unroll
      for (int j = 0; j < 4; ++j) kdst[j] = *(const uint4*)(krow + j * 8);
    }
#endif
    // ---- stage V tile transposed [vd][key] into LDS (manual) ----
    {
      const unsigned short* vrow =
          V + (size_t)(b * MK + kb + cr) * LD + h * 128 + cc;
      union { uint4 q[4]; unsigned short s[32]; } vv;
#pragma unroll
      for (int j = 0; j < 4; ++j) vv.q[j] = *(const uint4*)(vrow + j * 8);
#pragma unroll
      for (int j = 0; j < 32; ++j) Vt[(cc + j) * 32 + cr] = vv.s[j];

      if (kb + 32 < MK) { // prefetch next block (global_prefetch_b8)
        __builtin_prefetch(V + (size_t)(b * MK + kb + 32 + cr) * LD + h * 128 + cc, 0, 0);
      }
    }
#if HAVE_TDM
    if (wave == 0) __builtin_amdgcn_s_wait_tensorcnt(0);
#endif
    __syncthreads();

    // ---- S = Q K^T : two 16x16 tiles (keys kb..kb+15, kb+16..kb+31) ----
    v8f s0 = z, s1 = z;
#pragma unroll
    for (int c = 0; c < 4; ++c) {
      v16bf b0 = ld_b_bf16(&Kt[n * 128 + c * 32 + hf * 16]);
      s0 = WMMA_BF16(qf[c], b0, s0);
      v16bf b1 = ld_b_bf16(&Kt[(16 + n) * 128 + c * 32 + hf * 16]);
      s1 = WMMA_BF16(qf[c], b1, s1);
    }

    // ---- online softmax (rows live across the 16-lane half-groups) ----
    float p0[8], p1[8], alpha[8];
#pragma unroll
    for (int r = 0; r < 8; ++r) {
      float x0 = s0[r] * scale, x1 = s1[r] * scale;
      float mx = fmaxf(x0, x1);
      mx = fmaxf(mx, __shfl_xor(mx, 1));
      mx = fmaxf(mx, __shfl_xor(mx, 2));
      mx = fmaxf(mx, __shfl_xor(mx, 4));
      mx = fmaxf(mx, __shfl_xor(mx, 8));
      float mn = fmaxf(m[r], mx);
      float al = __expf(m[r] - mn);
      float e0 = __expf(x0 - mn), e1 = __expf(x1 - mn);
      float rs = e0 + e1;
      rs += __shfl_xor(rs, 1);
      rs += __shfl_xor(rs, 2);
      rs += __shfl_xor(rs, 4);
      rs += __shfl_xor(rs, 8);
      l[r] = l[r] * al + rs;
      m[r] = mn;
      alpha[r] = al;
      p0[r] = e0; p1[r] = e1;
    }
#pragma unroll
    for (int t = 0; t < 8; ++t) {
#pragma unroll
      for (int r = 0; r < 8; ++r) o[t][r] *= alpha[r];
    }

    // ---- repack P: C-layout -> A-layout via per-wave LDS scratch ----
    unsigned short* ps = Ps[wave];
#pragma unroll
    for (int r = 0; r < 8; ++r) {
      const int rr = r + 8 * hf;
      ps[rr * 32 + n]      = f2bf(p0[r]);
      ps[rr * 32 + 16 + n] = f2bf(p1[r]);
    }
    v16bf pa = ld_a_bf16(&ps[n * 32 + hf * 16], &ps[n * 32 + hf * 16 + 8]);

    // ---- O += P V (B-fragments contiguous thanks to transposed Vt) ----
#pragma unroll
    for (int t = 0; t < 8; ++t) {
      v16bf vb = ld_b_bf16(&Vt[(t * 16 + n) * 32 + hf * 16]);
      o[t] = WMMA_BF16(pa, vb, o[t]);
    }
  }

  // ---- epilogue: normalize and store bf16 ----
#pragma unroll
  for (int r = 0; r < 8; ++r) l[r] = 1.0f / l[r];
#pragma unroll
  for (int t = 0; t < 8; ++t) {
#pragma unroll
    for (int r = 0; r < 8; ++r) {
      const int orow = b * MQ + qb * 64 + wave * 16 + r + 8 * hf;
      O[(size_t)orow * LD + h * 128 + t * 16 + n] = f2bf(o[t][r] * l[r]);
    }
  }
}

// ---------------------------------------------------------------------------
// Output projection: Z[M,128](f32) = Y[M,1024](bf16) @ Wo[128,1024](bf16)^T + bo
// ---------------------------------------------------------------------------
__global__ __launch_bounds__(128)
void outproj_kernel(const unsigned short* __restrict__ Ypre,
                    const unsigned short* __restrict__ Wo,
                    const float* __restrict__ bo,
                    float* __restrict__ Z, int M) {
  const int tid  = threadIdx.x;
  const int wave = tid >> 5, lane = tid & 31;
  const int n = lane & 15, hf = lane >> 4;
  const int row  = blockIdx.x * 32 + (wave & 1) * 16 + n;
  const int col0 = (wave >> 1) * 64;
  (void)M;

  const unsigned short* yrow = Ypre + (size_t)row * LD;
  v8f z = {0.f, 0.f, 0.f, 0.f, 0.f, 0.f, 0.f, 0.f};
  v8f acc[4] = {z, z, z, z};

  for (int kc = 0; kc < 1024; kc += 32) {
    v16bf a = ld_a_bf16(yrow + kc + hf * 16, yrow + kc + hf * 16 + 8);
#pragma unroll
    for (int t = 0; t < 4; ++t) {
      const unsigned short* wrow = Wo + (size_t)(col0 + t * 16 + n) * 1024;
      v16bf bfr = ld_b_bf16(wrow + kc + hf * 16);
      acc[t] = WMMA_BF16(a, bfr, acc[t]);
    }
  }
#pragma unroll
  for (int t = 0; t < 4; ++t) {
    const int col = col0 + t * 16 + n;
    const float bv = bo[col];
#pragma unroll
    for (int r = 0; r < 8; ++r) {
      const int orow = blockIdx.x * 32 + (wave & 1) * 16 + r + 8 * hf;
      Z[(size_t)orow * 128 + col] = acc[t][r] + bv;
    }
  }
}

// ---------------------------------------------------------------------------
extern "C" void kernel_launch(void* const* d_in, const int* in_sizes, int n_in,
                              void* d_out, int out_size, void* d_ws, size_t ws_size,
                              hipStream_t stream) {
  (void)in_sizes; (void)n_in; (void)out_size; (void)ws_size;

  const float* queries = (const float*)d_in[0];
  const float* keys    = (const float*)d_in[1];
  const float* Wq  = (const float*)d_in[2];
  const float* bq  = (const float*)d_in[3];
  const float* Wk  = (const float*)d_in[4];
  const float* bk  = (const float*)d_in[5];
  const float* Wp  = (const float*)d_in[6];
  const float* bp  = (const float*)d_in[7];
  const float* Wg  = (const float*)d_in[8];
  const float* bg  = (const float*)d_in[9];
  const float* Wo1 = (const float*)d_in[10];
  const float* bo1 = (const float*)d_in[11];
  const float* Wo2 = (const float*)d_in[12];
  const float* bo2 = (const float*)d_in[13];

  const int MQ = BATCH * NQ_;  // 8192
  const int MK = BATCH * NK_;  // 16384
  const int NW = 1024 * 128;   // weight elements (each weight matrix)

  // workspace layout (bf16 as ushort)
  unsigned short* qb   = (unsigned short*)d_ws;                 // [8192,128]
  unsigned short* kbuf = qb   + (size_t)MQ * 128;               // [16384,128]
  unsigned short* wqb  = kbuf + (size_t)MK * 128;
  unsigned short* wkb  = wqb  + (size_t)NW;
  unsigned short* wpb  = wkb  + (size_t)NW;
  unsigned short* wgb  = wpb  + (size_t)NW;
  unsigned short* wo1b = wgb  + (size_t)NW;
  unsigned short* wo2b = wo1b + (size_t)NW;
  unsigned short* u    = wo2b + (size_t)NW;                     // [8192,1024]
  unsigned short* g    = u    + (size_t)MQ * LD;                // [8192,1024]
  unsigned short* v    = g    + (size_t)MQ * LD;                // [16384,1024]
  unsigned short* p    = v    + (size_t)MK * LD;                // [16384,1024]
  unsigned short* Ppre = p    + (size_t)MK * LD;                // [8192,1024]
  unsigned short* Dpre = Ppre + (size_t)MQ * LD;                // [16384,1024]

  float* Pout = (float*)d_out;                                  // [8192,128]
  float* Dout = Pout + (size_t)MQ * 128;                        // [16384,128]

  // 0) one-time f32 -> bf16 precast of activations and weights
  cvt_bf16_kernel<<<dim3((MQ * 128) / 2048), dim3(256), 0, stream>>>(queries, qb, MQ * 128);
  cvt_bf16_kernel<<<dim3((MK * 128) / 2048), dim3(256), 0, stream>>>(keys, kbuf, MK * 128);
  cvt_bf16_kernel<<<dim3(NW / 2048), dim3(256), 0, stream>>>(Wq,  wqb,  NW);
  cvt_bf16_kernel<<<dim3(NW / 2048), dim3(256), 0, stream>>>(Wk,  wkb,  NW);
  cvt_bf16_kernel<<<dim3(NW / 2048), dim3(256), 0, stream>>>(Wp,  wpb,  NW);
  cvt_bf16_kernel<<<dim3(NW / 2048), dim3(256), 0, stream>>>(Wg,  wgb,  NW);
  cvt_bf16_kernel<<<dim3(NW / 2048), dim3(256), 0, stream>>>(Wo1, wo1b, NW);
  cvt_bf16_kernel<<<dim3(NW / 2048), dim3(256), 0, stream>>>(Wo2, wo2b, NW);

  // 1) input projections (pure bf16 WMMA GEMMs)
  proj_kernel<<<dim3((MQ / 64) * 16), dim3(128), 0, stream>>>(qb,   wqb, bq, u, MQ);
  proj_kernel<<<dim3((MQ / 64) * 16), dim3(128), 0, stream>>>(qb,   wgb, bg, g, MQ);
  proj_kernel<<<dim3((MK / 64) * 16), dim3(128), 0, stream>>>(kbuf, wkb, bk, v, MK);
  proj_kernel<<<dim3((MK / 64) * 16), dim3(128), 0, stream>>>(kbuf, wpb, bp, p, MK);

  // 2) P branch: row softmax -> flash(Q=u, K=v, V=p)
  flash_kernel<<<dim3(BATCH * HEADS * (NQ_ / 64)), dim3(128), 0, stream>>>(
      u, v, p, Ppre, NQ_, NK_);
  // 3) D branch: column softmax == flash with swapped roles (Q=v, K=u, V=g)
  flash_kernel<<<dim3(BATCH * HEADS * (NK_ / 64)), dim3(128), 0, stream>>>(
      v, u, g, Dpre, NK_, NQ_);

  // 4) output projections (K=1024)
  outproj_kernel<<<dim3(MQ / 32), dim3(128), 0, stream>>>(Ppre, wo1b, bo1, Pout, MQ);
  outproj_kernel<<<dim3(MK / 32), dim3(128), 0, stream>>>(Dpre, wo2b, bo2, Dout, MK);
}
